// LearnableGaborFilter_5257039970368
// MI455X (gfx1250) — matrix operand order repaired and checked by reading, not compiled, not run on hardware
//
#include <hip/hip_runtime.h>
#include <hip/hip_bf16.h>
#include <stdint.h>

typedef __attribute__((ext_vector_type(16))) _Float16 v16h;
typedef __attribute__((ext_vector_type(4)))  _Float16 v4h;
typedef __attribute__((ext_vector_type(8)))  float    v8f;
typedef __attribute__((ext_vector_type(4)))  unsigned int v4u;
typedef __attribute__((ext_vector_type(8)))  int      v8i;
typedef __attribute__((ext_vector_type(4)))  int      v4i;

#define IMG   256
#define NIMG  64
#define OCH   8
#define KS    15
#define TW    32    // output cols per block
#define THT   16    // output rows per block
#define LW    48    // LDS tile width  = TW + 16
#define LH    32    // LDS tile height = THT + 16
#define PAD   8     // halo pad in padded xs buffer
#define WPAD  280   // padded row stride (>= 273, multiple of 8)
#define HPAD  280   // padded rows per image

// ---------------------------------------------------------------------------
// Kernel 1: build the 8 Gabor filters, normalize (zero mean, std ddof=1),
// emit f16 in V_WMMA_F32_16X16X32_F16 B-operand layout:
//   b_buf[(s*32 + lane)*16 + h], N = lane%16, ky = 2*s + (lane>=16), kx = h;
//   zero for N>=8 / ky==15 / kx==15 (K padded 225 -> 256).
// ---------------------------------------------------------------------------
__global__ void gabor_prep(const float* __restrict__ theta,
                           const float* __restrict__ sigma,
                           const float* __restrict__ lambd,
                           const float* __restrict__ gamma,
                           const float* __restrict__ psi,
                           _Float16* __restrict__ b_buf) {
    __shared__ float filt[OCH][KS][KS];
    const int t = threadIdx.x;
    const int w = t >> 5;        // wave id = filter id
    const int lane = t & 31;

    const float th = theta[w];
    const float sg = fmaxf(sigma[w], 1.0f);
    const float lb = fmaxf(lambd[w], 2.0f);
    const float gm = fminf(fmaxf(gamma[w], 0.1f), 1.0f);
    const float ps = psi[w];
    const float ct = cosf(th), st = sinf(th);
    const float inv_sg2 = 1.0f / (sg * sg);
    const float gm2 = gm * gm;
    const float twoPi = 6.28318530717958647692f;

    float vals[8];
    float lsum = 0.0f;
    #pragma unroll
    for (int k = 0; k < 8; ++k) {
        const int tap = lane + k * 32;
        float v = 0.0f;
        if (tap < KS * KS) {
            const int i = tap / KS, j = tap % KS;       // (ky, kx)
            const float yg = (float)(i - 7);
            const float xg = (float)(j - 7);
            const float x_th =  xg * ct + yg * st;
            const float y_th = -xg * st + yg * ct;
            const float gauss = expf(-0.5f * (x_th * x_th + gm2 * y_th * y_th) * inv_sg2);
            const float sinus = cosf(twoPi * x_th / lb + ps);
            v = gauss * sinus;
            lsum += v;
        }
        vals[k] = v;
    }
    #pragma unroll
    for (int off = 16; off; off >>= 1) lsum += __shfl_xor(lsum, off, 32);
    const float mean = lsum / 225.0f;

    float lvar = 0.0f;
    #pragma unroll
    for (int k = 0; k < 8; ++k) {
        const int tap = lane + k * 32;
        if (tap < KS * KS) { const float d = vals[k] - mean; lvar += d * d; }
    }
    #pragma unroll
    for (int off = 16; off; off >>= 1) lvar += __shfl_xor(lvar, off, 32);
    const float inv_std = 1.0f / (sqrtf(lvar / 224.0f) + 1e-8f);

    #pragma unroll
    for (int k = 0; k < 8; ++k) {
        const int tap = lane + k * 32;
        if (tap < KS * KS)
            filt[w][tap / KS][tap % KS] = (vals[k] - mean) * inv_std;
    }
    __syncthreads();

    for (int e = t; e < 8 * 32 * 16; e += 256) {
        const int s  = e >> 9;
        const int L  = (e >> 4) & 31;
        const int h  = e & 15;
        const int N  = L & 15;
        const int ky = 2 * s + (L >= 16 ? 1 : 0);
        const int kx = h;
        float v = 0.0f;
        if (N < OCH && ky < KS && kx < KS) v = filt[N][ky][kx];
        b_buf[e] = (_Float16)v;
    }
}

// ---------------------------------------------------------------------------
// Kernel 2a: zero the padded xs buffer (halo ring must be 0 for the conv).
// ---------------------------------------------------------------------------
__global__ void zero_pad(_Float16* __restrict__ xsp) {
    const long i = ((long)blockIdx.x * 256 + threadIdx.x) * 8;
    const float4 z = {0.f, 0.f, 0.f, 0.f};
    *(float4*)(xsp + i) = z;
}

// ---------------------------------------------------------------------------
// Kernel 2b: xs = sum over 3 channels -> f16 into padded buffer interior.
// ---------------------------------------------------------------------------
__global__ void chsum(const float* __restrict__ x, _Float16* __restrict__ xsp) {
    const long idx  = (long)blockIdx.x * blockDim.x + threadIdx.x;
    const long base = idx * 4;                  // 4 consecutive pixels
    const long n    = base >> 16;
    const long pix  = base & 65535;
    const int  y    = (int)(pix >> 8);
    const int  xc   = (int)(pix & 255);
    const float4 a = *(const float4*)(x + (n * 3 + 0) * 65536 + pix);
    const float4 b = *(const float4*)(x + (n * 3 + 1) * 65536 + pix);
    const float4 c = *(const float4*)(x + (n * 3 + 2) * 65536 + pix);
    v4h r;
    r[0] = (_Float16)(a.x + b.x + c.x);
    r[1] = (_Float16)(a.y + b.y + c.y);
    r[2] = (_Float16)(a.z + b.z + c.z);
    r[3] = (_Float16)(a.w + b.w + c.w);
    *(v4h*)(xsp + (n * HPAD + y + PAD) * (long)WPAD + xc + PAD) = r;
}

// ---------------------------------------------------------------------------
// Kernel 3: implicit-GEMM conv via v_wmma_f32_16x16x32_f16.
// Global->LDS staging done by the Tensor Data Mover (one tensor_load_to_lds
// per block, issued by wave 0; 48x32 f16 tile, stride 280).
// ---------------------------------------------------------------------------
__global__ void __launch_bounds__(256)
gabor_conv(const _Float16* __restrict__ xsp,
           const _Float16* __restrict__ b_buf,
           float* __restrict__ out) {
    __shared__ _Float16 stile[LH * LW];

    const int lane   = threadIdx.x & 31;
    const int wv     = threadIdx.x >> 5;
    const int laneM  = lane & 15;
    const int hiLane = (lane >= 16) ? 1 : 0;

    const int C0 = blockIdx.x * TW;
    const int R0 = blockIdx.y * THT;
    const int n  = blockIdx.z;

    // B operand: constant across the grid; hold in registers (8 x v16h).
    v16h breg[8];
    const v16h* Bp = (const v16h*)b_buf;
    #pragma unroll
    for (int s = 0; s < 8; ++s) breg[s] = Bp[s * 32 + lane];

    // --- TDM: DMA the 48x32 halo tile into LDS (wave 0 only) ---------------
    if (wv == 0) {
        // Tile origin in padded coords: row R0-7+PAD = R0+1, col C0+1.
        const unsigned long long ga =
            (unsigned long long)(uintptr_t)xsp +
            2ull * (((unsigned long long)n * HPAD + (unsigned)(R0 + 1)) * WPAD
                    + (unsigned)(C0 + 1));
        const unsigned ldsoff = (unsigned)(uintptr_t)stile;  // low 32b = LDS offset

        v4u g0;
        g0[0] = 1u;                                   // count=1, no gather
        g0[1] = ldsoff;                               // lds_addr
        g0[2] = (unsigned)ga;                         // global_addr[31:0]
        g0[3] = (unsigned)((ga >> 32) & 0x1FFFFFFu)   // global_addr[56:32]
              | (2u << 30);                           // type = 2 (image)
        v8i g1;
        g1[0] = 0x10000;                              // data_size=1 (2 bytes)
        g1[1] = (int)((unsigned)WPAD << 16);          // tensor_dim0 = 280
        g1[2] = (int)((unsigned)(NIMG * HPAD) << 16); // tensor_dim1 = 17920
        g1[3] = (int)(48u << 16);                     // tile_dim0 = 48
        g1[4] = 32;                                   // tile_dim1 = 32
        g1[5] = WPAD;                                 // tensor_dim0_stride = 280
        g1[6] = 0;
        g1[7] = 0;
        v4i gz4 = {0, 0, 0, 0};                       // groups 2/3 unused (2-D)
        v8i gz8 = {0, 0, 0, 0, 0, 0, 0, 0};          // trailing group (unused)

        __builtin_amdgcn_tensor_load_to_lds(g0, g1, gz4, gz4, gz8, 0);
        __builtin_amdgcn_s_wait_tensorcnt(0);
    }
    __syncthreads();

    // 32 wave-tiles (16 rows x 2 col-segments); wave wv owns 4 of them.
    #pragma unroll
    for (int q = 0; q < 4; ++q) {
        const int mt  = wv * 4 + q;
        const int tr  = mt >> 1;
        const int seg = (mt & 1) * 16;
        const int colbase = seg + laneM;

        // Preload all 8 K-step A fragments, then run the WMMA chain
        // back-to-back on the f32 accumulator.
        v16h a[8];
        #pragma unroll
        for (int s = 0; s < 8; ++s) {
            const _Float16* row0 = &stile[(tr + 2 * s + hiLane) * LW + colbase];
            #pragma unroll
            for (int j = 0; j < 16; ++j) { a[s][j] = row0[j]; }
        }

        v8f acc = {};
        #pragma unroll
        for (int s = 0; s < 8; ++s)
            acc = __builtin_amdgcn_wmma_f32_16x16x32_f16(
                false, a[s], false, breg[s], (short)0, acc, false, false);

        const int ch = laneM;
        if (ch < OCH) {
            float* p = out + (((long)n * OCH + ch) * IMG + (R0 + tr)) * IMG
                           + C0 + seg + hiLane * 8;
            float4 lo = {acc[0], acc[1], acc[2], acc[3]};
            float4 hi = {acc[4], acc[5], acc[6], acc[7]};
            *(float4*)(p)     = lo;
            *(float4*)(p + 4) = hi;
        }
    }
}

// ---------------------------------------------------------------------------
// ws layout: [0, 8KB) B operand | [8KB, 8KB + 64*280*280*2) padded f16 xs.
// ---------------------------------------------------------------------------
extern "C" void kernel_launch(void* const* d_in, const int* in_sizes, int n_in,
                              void* d_out, int out_size, void* d_ws, size_t ws_size,
                              hipStream_t stream) {
    const float* x     = (const float*)d_in[0];
    const float* theta = (const float*)d_in[1];
    const float* sigma = (const float*)d_in[2];
    const float* lambd = (const float*)d_in[3];
    const float* gamma = (const float*)d_in[4];
    const float* psi   = (const float*)d_in[5];
    float* out = (float*)d_out;

    _Float16* b_buf = (_Float16*)d_ws;
    _Float16* xsp   = (_Float16*)((char*)d_ws + 8192);

    const long npad = (long)NIMG * HPAD * WPAD;          // 5,017,600 halfs
    zero_pad<<<(int)(npad / 8 / 256), 256, 0, stream>>>(xsp);

    gabor_prep<<<1, 256, 0, stream>>>(theta, sigma, lambd, gamma, psi, b_buf);

    const long npix = (long)NIMG * IMG * IMG;            // 4,194,304
    chsum<<<(int)(npix / 4 / 256), 256, 0, stream>>>(x, xsp);

    dim3 grid(IMG / TW, IMG / THT, NIMG);                // 8 x 16 x 64
    gabor_conv<<<grid, 256, 0, stream>>>(xsp, b_buf, out);
}
